// AutoCorrelation_69793218560064
// MI455X (gfx1250) — matrix operand order repaired and checked by reading, not compiled, not run on hardware
//
#include <hip/hip_runtime.h>

// Problem constants (B,L,H,E) = (16,2048,8,64); k = int(log(2048)) = 7.
#define LL      2048
#define CH      512        // H*E, contiguous per (b,l)
#define NB      16
#define TOPK    7
#define TM      128
#define TN      128
#define KC      32         // WMMA K per step (bf16 16x16x32)
#define LSTRIDE 40         // bf16 row stride of staged tiles (80B: 16B aligned, conflict-free)
#define GSTRIDE 129        // float row stride of Gram tile

typedef __bf16 v4bf  __attribute__((ext_vector_type(4)));
typedef __bf16 v8bf  __attribute__((ext_vector_type(8)));
typedef __bf16 v16bf __attribute__((ext_vector_type(16)));
typedef float  v8f   __attribute__((ext_vector_type(8)));

__device__ __forceinline__ void cvt4_hilo(float4 f, v4bf& hi, v4bf& lo) {
  float a[4] = {f.x, f.y, f.z, f.w};
#pragma unroll
  for (int j = 0; j < 4; ++j) {
    __bf16 h = (__bf16)a[j];          // RNE truncate f32 -> bf16
    hi[j] = h;
    lo[j] = (__bf16)(a[j] - (float)h); // residual, captures next 8 mantissa bits
  }
}

__device__ __forceinline__ v16bf cat8(v8bf a, v8bf b) {
  return __builtin_shufflevector(a, b, 0,1,2,3,4,5,6,7,8,9,10,11,12,13,14,15);
}

__global__ void zero_f32(float* p, int n) {
  int i = blockIdx.x * blockDim.x + threadIdx.x;
  if (i < n) p[i] = 0.0f;
}

// Batched Q·Kᵀ (TN gemm) with circular-diagonal reduction epilogue.
// Grid: (L/TM, L/TN, B). Block: 256 threads = 8 waves (2 M-waves x 4 N-waves).
__global__ __launch_bounds__(256) void corr_gemm_diag(
    const float* __restrict__ q, const float* __restrict__ kmat,
    float* __restrict__ meanr) {
  // LDS: staging (Qhi/Qlo/Khi/Klo, 4*10240 B) aliased under the 128x129 f32 Gram tile.
  __shared__ __align__(16) unsigned char smem[TM * GSTRIDE * 4]; // 66048 B
  __bf16* Qhi = (__bf16*)(smem +     0);
  __bf16* Qlo = (__bf16*)(smem + 10240);
  __bf16* Khi = (__bf16*)(smem + 20480);
  __bf16* Klo = (__bf16*)(smem + 30720);
  float*  Gs  = (float*)smem;

  const int tid  = threadIdx.x;
  const int lane = tid & 31;
  const int wid  = tid >> 5;
  const int wm   = wid & 1;          // wave M tile: 64 rows
  const int wn   = wid >> 1;         // wave N tile: 32 cols
  const int m0   = blockIdx.x * TM;
  const int n0   = blockIdx.y * TN;
  const int b    = blockIdx.z;
  const float* qb = q    + (size_t)b * LL * CH;
  const float* kb = kmat + (size_t)b * LL * CH;

  const int h  = lane >> 4;          // lane half (selects K sub-block per ISA layout)
  const int mr = lane & 15;          // row-within-tile / n-within-tile

  v8f acc[4][2] = {};

  for (int k0 = 0; k0 < CH; k0 += KC) {
    __syncthreads();
    // Stage 128x32 f32 tiles of Q and K as hi/lo bf16 (coalesced float4 reads).
#pragma unroll
    for (int i = 0; i < 4; ++i) {
      int idx = tid + i * 256;       // float4 index 0..1023
      int row = idx >> 3;            // 8 float4 per 32-float row
      int col = (idx & 7) << 2;
      float4 qv = *(const float4*)(qb + (size_t)(m0 + row) * CH + k0 + col);
      float4 kv = *(const float4*)(kb + (size_t)(n0 + row) * CH + k0 + col);
      v4bf qh, ql, kh, kl;
      cvt4_hilo(qv, qh, ql);
      cvt4_hilo(kv, kh, kl);
      *(v4bf*)(Qhi + row * LSTRIDE + col) = qh;
      *(v4bf*)(Qlo + row * LSTRIDE + col) = ql;
      *(v4bf*)(Khi + row * LSTRIDE + col) = kh;
      *(v4bf*)(Klo + row * LSTRIDE + col) = kl;
    }
    __syncthreads();

    // A (16x32 bf16): lane {m=mr}, k = 16h..16h+15 packed sequentially in v0..7.
    v16bf aH[4], aL[4], bH[2], bL[2];
#pragma unroll
    for (int sm = 0; sm < 4; ++sm) {
      const __bf16* ph = Qhi + (wm * 64 + sm * 16 + mr) * LSTRIDE;
      const __bf16* pl = Qlo + (wm * 64 + sm * 16 + mr) * LSTRIDE;
      aH[sm] = cat8(*(const v8bf*)(ph + 16 * h), *(const v8bf*)(ph + 16 * h + 8));
      aL[sm] = cat8(*(const v8bf*)(pl + 16 * h), *(const v8bf*)(pl + 16 * h + 8));
    }
    // B (32x16 bf16): lane {n=mr}, k = 16h..16h+15 packed sequentially in v0..7.
#pragma unroll
    for (int sn = 0; sn < 2; ++sn) {
      const __bf16* ph = Khi + (wn * 32 + sn * 16 + mr) * LSTRIDE + 16 * h;
      const __bf16* pl = Klo + (wn * 32 + sn * 16 + mr) * LSTRIDE + 16 * h;
      bH[sn] = cat8(*(const v8bf*)(ph), *(const v8bf*)(ph + 8));
      bL[sn] = cat8(*(const v8bf*)(pl), *(const v8bf*)(pl + 8));
    }
    // Split-precision accumulate: hi*hi + hi*lo + lo*hi (fp32 accumulators).
#pragma unroll
    for (int sm = 0; sm < 4; ++sm)
#pragma unroll
      for (int sn = 0; sn < 2; ++sn) {
        acc[sm][sn] = __builtin_amdgcn_wmma_f32_16x16x32_bf16(
            false, aH[sm], false, bH[sn], (short)0, acc[sm][sn], false, false);
        acc[sm][sn] = __builtin_amdgcn_wmma_f32_16x16x32_bf16(
            false, aH[sm], false, bL[sn], (short)0, acc[sm][sn], false, false);
        acc[sm][sn] = __builtin_amdgcn_wmma_f32_16x16x32_bf16(
            false, aL[sm], false, bH[sn], (short)0, acc[sm][sn], false, false);
      }
  }

  // Spill C tile to LDS Gram buffer (C layout: lane<16 -> n=lane,m=r; lane>=16 -> n=lane-16,m=8+r).
  __syncthreads();
#pragma unroll
  for (int sm = 0; sm < 4; ++sm)
#pragma unroll
    for (int sn = 0; sn < 2; ++sn)
#pragma unroll
      for (int r = 0; r < 8; ++r) {
        int ml = wm * 64 + sm * 16 + h * 8 + r;
        int nl = wn * 32 + sn * 16 + mr;
        Gs[ml * GSTRIDE + nl] = acc[sm][sn][r];
      }
  __syncthreads();

  // Reduce the 255 circular diagonals; one global atomic per diagonal.
  for (int d = tid; d < 255; d += 256) {
    int off = d - 127;                       // t - s within tile
    int t   = off > 0 ? off : 0;
    int s   = t - off;
    int cnt = 128 - (off < 0 ? -off : off);
    const float* p = Gs + t * GSTRIDE + s;
    float sum = 0.0f;
    for (int i = 0; i < cnt; ++i) { sum += *p; p += GSTRIDE + 1; }
    int tau = (m0 - n0 + off + 4096) & (LL - 1);
    atomicAdd(meanr + b * LL + tau, sum);
  }
}

// Per-batch top-7 (lowest-index tie-break, like jax.lax.top_k) + softmax.
__global__ __launch_bounds__(256) void topk_softmax(
    const float* __restrict__ meanr, float* __restrict__ wout, int* __restrict__ tout) {
  __shared__ float vals[LL];
  __shared__ float rv[256];
  __shared__ int   ri[256];
  __shared__ float topv[TOPK];
  __shared__ int   topi[TOPK];
  const int b = blockIdx.x, tid = threadIdx.x;
  for (int i = tid; i < LL; i += 256) vals[i] = meanr[b * LL + i] * (1.0f / 512.0f);
  __syncthreads();
  for (int it = 0; it < TOPK; ++it) {
    float best = -3.4e38f; int bi = 0;
    for (int i = tid; i < LL; i += 256) {
      float x = vals[i];
      if (x > best) { best = x; bi = i; }
    }
    rv[tid] = best; ri[tid] = bi;
    __syncthreads();
    for (int s = 128; s > 0; s >>= 1) {
      if (tid < s) {
        if (rv[tid + s] > rv[tid] || (rv[tid + s] == rv[tid] && ri[tid + s] < ri[tid])) {
          rv[tid] = rv[tid + s]; ri[tid] = ri[tid + s];
        }
      }
      __syncthreads();
    }
    if (tid == 0) { topv[it] = rv[0]; topi[it] = ri[0]; vals[ri[0]] = -3.4e38f; }
    __syncthreads();
  }
  if (tid == 0) {
    float m = topv[0], ssum = 0.0f, e[TOPK];
    for (int j = 0; j < TOPK; ++j) { e[j] = __expf(topv[j] - m); ssum += e[j]; }
    for (int j = 0; j < TOPK; ++j) {
      wout[b * 8 + j] = e[j] / ssum;
      tout[b * 8 + j] = topi[j];
    }
  }
}

// out[b,l,:] = Σ_j w[b,j] · v[b,(l+tao_j) mod L,:]   (float4 vectorized)
__global__ __launch_bounds__(256) void gather_agg(
    const float* __restrict__ v, const float* __restrict__ w,
    const int* __restrict__ tao, float* __restrict__ out) {
  int idx = blockIdx.x * 256 + threadIdx.x;   // float4 unit index
  int c4  = idx & 127;                        // 512/4 chunks
  int l   = (idx >> 7) & (LL - 1);
  int b   = idx >> 18;
  float4 a; a.x = a.y = a.z = a.w = 0.0f;
#pragma unroll
  for (int j = 0; j < TOPK; ++j) {
    float wj = w[b * 8 + j];
    int src  = (l + tao[b * 8 + j]) & (LL - 1);
    float4 vv = *(const float4*)(v + (size_t)(b * LL + src) * CH + (c4 << 2));
    a.x += wj * vv.x; a.y += wj * vv.y; a.z += wj * vv.z; a.w += wj * vv.w;
  }
  *(float4*)(out + (size_t)idx * 4) = a;
}

extern "C" void kernel_launch(void* const* d_in, const int* in_sizes, int n_in,
                              void* d_out, int out_size, void* d_ws, size_t ws_size,
                              hipStream_t stream) {
  const float* q = (const float*)d_in[0];
  const float* k = (const float*)d_in[1];
  const float* v = (const float*)d_in[2];
  float* out = (float*)d_out;

  // Workspace layout: meanr[16*2048] f32 | weights[16*8] f32 | tao[16*8] i32
  float* meanr = (float*)d_ws;
  float* wts   = (float*)((char*)d_ws + (size_t)NB * LL * sizeof(float));
  int*   tao   = (int*)((char*)d_ws + (size_t)NB * LL * sizeof(float) + NB * 8 * sizeof(float));

  zero_f32<<<(NB * LL + 255) / 256, 256, 0, stream>>>(meanr, NB * LL);
  corr_gemm_diag<<<dim3(LL / TM, LL / TN, NB), 256, 0, stream>>>(q, k, meanr);
  topk_softmax<<<NB, 256, 0, stream>>>(meanr, wts, tao);
  gather_agg<<<(NB * LL * (CH / 4)) / 256, 256, 0, stream>>>(v, wts, tao, out);
}
